// Point_contextual_hyper_prior_decoder_6047313953102
// MI455X (gfx1250) — matrix operand order repaired and checked by reading, not compiled, not run on hardware
//
#include <hip/hip_runtime.h>

// ---------------------------------------------------------------------------
// CDNA5 (gfx1250, wave32) sparse generative conv decoder.
//  - dense 128^3 voxel->row map (8MB, L2 resident)
//  - bf16 activations (61MB feature map, L2 resident)
//  - all matmuls via v_wmma_f32_16x16x32_bf16 (fully unrolled: 8 WMMA/offset)
//  - conv layers: 1024-thread blocks stage the full 221KB transposed bf16
//    weight set into LDS via the Tensor Data Mover (tensor_load_to_lds,
//    TENSORcnt-tracked, zero VGPR traffic); 32 waves/block share it.
//    One wave owns a 16-point x 64-channel tile: 4 f32 accumulators, A
//    fragments (neighbor gathers) reused across the 4 output-channel tiles.
// ---------------------------------------------------------------------------

typedef __attribute__((ext_vector_type(16))) __bf16 v16bf;
typedef __attribute__((ext_vector_type(8)))  float  v8f;
typedef __attribute__((ext_vector_type(4)))  unsigned int u32x4;
typedef __attribute__((ext_vector_type(8)))  unsigned int u32x8;

union FragA { v16bf v; u32x4 q[2]; };

#define WMMA_BF16(A, B, C) \
  __builtin_amdgcn_wmma_f32_16x16x32_bf16(false, (A), false, (B), (short)0, (C), false, false)

#define CONV_LDS_BYTES (27 * 64 * 64 * 2)  // 221184 B, fits 320KB WGP LDS
#define USE_TDM 1

// ------------------------------ helpers ------------------------------------

__global__ void k_init_map(int* __restrict__ map, int n) {
  int i = blockIdx.x * blockDim.x + threadIdx.x;
  if (i < n) map[i] = -1;
}

__global__ void k_build_map(const int* __restrict__ coords, int* __restrict__ map, int npts) {
  int i = blockIdx.x * blockDim.x + threadIdx.x;
  if (i >= npts) return;
  int n = i >> 3, k = i & 7;
  int cx = 2 * coords[3 * n + 0] + ((k >> 2) & 1);
  int cy = 2 * coords[3 * n + 1] + ((k >> 1) & 1);
  int cz = 2 * coords[3 * n + 2] + (k & 1);
  map[(((cx << 7) + cy) << 7) + cz] = i;
}

__global__ void k_f32_to_bf16(const float* __restrict__ src, __bf16* __restrict__ dst, int n) {
  int i = blockIdx.x * blockDim.x + threadIdx.x;
  if (i < n) dst[i] = (__bf16)src[i];
}

// Convert W[k][c][d] (f32) -> Wt[k][d][c] (bf16); one block per 64x64 matrix.
__global__ void k_weight_tr(const float* __restrict__ W, __bf16* __restrict__ Wt) {
  int k = blockIdx.x;
  const float* src = W + (size_t)k * 4096;
  __bf16* dst = Wt + (size_t)k * 4096;
  for (int e = threadIdx.x; e < 4096; e += blockDim.x) {
    int c = e >> 6, d = e & 63;
    dst[d * 64 + c] = (__bf16)src[c * 64 + d];
  }
}

// ------------------------------ upsample -----------------------------------
// x0[n*8+k][d] = relu( sum_c feats[n][c] * Wup[k][c][d] + b[d] )
// Wave per (16-row tile, child k). A loaded once, reused over 4 N-tiles.
__global__ void k_upsample(const __bf16* __restrict__ featsbf,
                           const __bf16* __restrict__ Wupt,
                           const float* __restrict__ bias,
                           __bf16* __restrict__ xout, int n_in) {
  int wave = (blockIdx.x * blockDim.x + threadIdx.x) >> 5;
  int lane = threadIdx.x & 31;
  int ntile = (n_in + 15) >> 4;
  if (wave >= ntile * 8) return;
  int tile = wave >> 3, k = wave & 7;
  int m = lane & 15, kh = lane >> 4;
  bool full = (tile + 1) * 16 <= n_in;  // wave-uniform

  int row = tile * 16 + m;
  int rowc = row < n_in ? row : n_in - 1;
  const __bf16* rp = featsbf + (size_t)rowc * 64;
  FragA a0, a1;
  a0.q[0] = *(const u32x4*)(rp + kh * 8);
  a0.q[1] = *(const u32x4*)(rp + 16 + kh * 8);
  a1.q[0] = *(const u32x4*)(rp + 32 + kh * 8);
  a1.q[1] = *(const u32x4*)(rp + 48 + kh * 8);

  const __bf16* wbase = Wupt + (size_t)k * 4096;
#pragma unroll
  for (int nt = 0; nt < 4; ++nt) {
    int dcol = nt * 16 + m;
    const __bf16* wp = wbase + dcol * 64 + kh * 8;
    FragA b0, b1;
    b0.q[0] = *(const u32x4*)(wp);
    b0.q[1] = *(const u32x4*)(wp + 16);
    b1.q[0] = *(const u32x4*)(wp + 32);
    b1.q[1] = *(const u32x4*)(wp + 48);
    v8f acc = {};
    acc = WMMA_BF16(a0.v, b0.v, acc);
    acc = WMMA_BF16(a1.v, b1.v, acc);
    float bv = bias[dcol];
    __bf16* op = xout + ((size_t)(tile * 16 + kh * 8) * 8 + k) * 64 + dcol;
    if (full) {
#pragma unroll
      for (int r = 0; r < 8; ++r) {
        float v = acc[r] + bv;
        v = v > 0.f ? v : 0.f;
        op[(size_t)r * 8 * 64] = (__bf16)v;
      }
    } else {
#pragma unroll
      for (int r = 0; r < 8; ++r) {
        if (tile * 16 + kh * 8 + r < n_in) {
          float v = acc[r] + bv;
          v = v > 0.f ? v : 0.f;
          op[(size_t)r * 8 * 64] = (__bf16)v;
        }
      }
    }
  }
}

// ------------------------------ sparse conv --------------------------------
// out[p][d] = sum_{off in 27} xin[map[coord(p)+off]] @ Wt[off] + b[d]
// 1024-thread block: TDM-stages all 27 weight matrices (bf16, transposed)
// into LDS, then 32 waves each own one 16-point tile (4 accumulators = 64ch).
template <bool RELU, bool OUTF32>
__global__ void __launch_bounds__(1024)
k_conv(const __bf16* __restrict__ xin,
       const __bf16* __restrict__ Wt,
       const float* __restrict__ bias,
       const int* __restrict__ map,
       const int* __restrict__ coords,
       __bf16* __restrict__ outbf, float* __restrict__ outf,
       int npts) {
  extern __shared__ char smem_raw[];
  __bf16* Wl = (__bf16*)smem_raw;

#if USE_TDM
  // Tensor Data Mover: one DMA of the whole 221184B weight set into LDS.
  // D# group0: count=1 | lds_addr=0 | global_addr | type=2.
  // D# group1: data_size=3 (8B), tensor 1024x27 (8B units), tile 1024x27,
  //            dim0_stride=1024 -> fully contiguous 2D copy.
  if ((threadIdx.x >> 5) == 0) {  // wave 0 only (TDM ignores EXEC; 1 issue/wave)
    unsigned long long ga = (unsigned long long)Wt;
    u32x4 g0 = { 1u, 0u, (unsigned)ga,
                 (unsigned)((ga >> 32) & 0x01FFFFFFu) | (2u << 30) };
    u32x8 g1 = { 0x00030000u,        // workgroup_mask=0, data_size=3 (8B)
                 1024u << 16,        // tensor_dim0 = 1024 (low 16 bits)
                 27u << 16,          // tensor_dim0 hi=0 | tensor_dim1 = 27
                 1024u << 16,        // tensor_dim1 hi=0 | tile_dim0 = 1024
                 27u,                // tile_dim1 = 27, tile_dim2 = 0
                 1024u,              // tensor_dim0_stride lo
                 0u, 0u };           // stride hi, tensor_dim1_stride (unused)
    asm volatile("tensor_load_to_lds %0, %1" :: "s"(g0), "s"(g1) : "memory");
    __builtin_amdgcn_s_wait_tensorcnt(0);
  }
#else
  {
    u32x4* d = (u32x4*)smem_raw;
    const u32x4* s = (const u32x4*)Wt;
    const int nchunks = CONV_LDS_BYTES / 16;
    for (int i = threadIdx.x; i < nchunks; i += 1024) d[i] = s[i];
  }
#endif
  __syncthreads();

  int tile = blockIdx.x * 32 + (threadIdx.x >> 5);
  int lane = threadIdx.x & 31;
  int ntile = (npts + 15) >> 4;
  if (tile < ntile) {
    int m = lane & 15, kh = lane >> 4;
    int p = tile * 16 + m;
    if (p >= npts) p = npts - 1;  // garbage rows only pollute rows never stored
    int n = p >> 3, k = p & 7;
    int cx = 2 * coords[3 * n + 0] + ((k >> 2) & 1);
    int cy = 2 * coords[3 * n + 1] + ((k >> 1) & 1);
    int cz = 2 * coords[3 * n + 2] + (k & 1);

    v8f acc0 = {}, acc1 = {}, acc2 = {}, acc3 = {};
    const u32x4 zz = {0u, 0u, 0u, 0u};

    for (int ox = -1; ox <= 1; ++ox) {
      for (int oy = -1; oy <= 1; ++oy) {
        int qx = cx + ox, qy = cy + oy;
        bool inxy = ((unsigned)qx < 128u) & ((unsigned)qy < 128u);
        int rowxy = ((((qx & 127) << 7) + (qy & 127)) << 7);
        int offb = ((ox + 1) * 3 + (oy + 1)) * 3;
#pragma unroll
        for (int oz = -1; oz <= 1; ++oz) {
          int qz = cz + oz;
          bool inb = inxy & ((unsigned)qz < 128u);
          int idx = inb ? map[rowxy + (qz & 127)] : -1;
          bool found = idx >= 0;
#if defined(__has_builtin)
#if __has_builtin(__builtin_amdgcn_ballot_w32)
          if (__builtin_amdgcn_ballot_w32(found) == 0u) continue;  // uniform skip
#endif
#endif
          const __bf16* rp = xin + (size_t)(found ? idx : 0) * 64;
          FragA a0, a1;
          a0.q[0] = found ? *(const u32x4*)(rp + kh * 8) : zz;
          a0.q[1] = found ? *(const u32x4*)(rp + 16 + kh * 8) : zz;
          a1.q[0] = found ? *(const u32x4*)(rp + 32 + kh * 8) : zz;
          a1.q[1] = found ? *(const u32x4*)(rp + 48 + kh * 8) : zz;

          const __bf16* wo = Wl + (offb + (oz + 1)) * 4096;
#pragma unroll
          for (int nt = 0; nt < 4; ++nt) {
            const __bf16* wp = wo + (nt * 16 + m) * 64 + kh * 8;
            FragA b0, b1;
            b0.q[0] = *(const u32x4*)(wp);
            b0.q[1] = *(const u32x4*)(wp + 16);
            b1.q[0] = *(const u32x4*)(wp + 32);
            b1.q[1] = *(const u32x4*)(wp + 48);
            v8f& acc = nt == 0 ? acc0 : nt == 1 ? acc1 : nt == 2 ? acc2 : acc3;
            acc = WMMA_BF16(a0.v, b0.v, acc);
            acc = WMMA_BF16(a1.v, b1.v, acc);
          }
        }
      }
    }

    bool full = (tile + 1) * 16 <= npts;  // wave-uniform
#pragma unroll
    for (int nt = 0; nt < 4; ++nt) {
      int dcol = nt * 16 + m;
      float bv = bias[dcol];
      v8f& acc = nt == 0 ? acc0 : nt == 1 ? acc1 : nt == 2 ? acc2 : acc3;
      size_t base = (size_t)(tile * 16 + kh * 8) * 64 + dcol;
      if (full) {
#pragma unroll
        for (int r = 0; r < 8; ++r) {
          float v = acc[r] + bv;
          if (RELU) v = v > 0.f ? v : 0.f;
          if (OUTF32) outf[base + (size_t)r * 64] = v;
          else        outbf[base + (size_t)r * 64] = (__bf16)v;
        }
      } else {
#pragma unroll
        for (int r = 0; r < 8; ++r) {
          if (tile * 16 + kh * 8 + r < npts) {
            float v = acc[r] + bv;
            if (RELU) v = v > 0.f ? v : 0.f;
            if (OUTF32) outf[base + (size_t)r * 64] = v;
            else        outbf[base + (size_t)r * 64] = (__bf16)v;
          }
        }
      }
    }
  }
}

// ------------------------------ launcher -----------------------------------

extern "C" void kernel_launch(void* const* d_in, const int* in_sizes, int n_in_cnt,
                              void* d_out, int out_size, void* d_ws, size_t ws_size,
                              hipStream_t stream) {
  const int*   coords = (const int*)d_in[0];
  const float* feats  = (const float*)d_in[1];
  const float* W_up   = (const float*)d_in[2];
  const float* b_up   = (const float*)d_in[3];
  const float* W1     = (const float*)d_in[4];
  const float* b1     = (const float*)d_in[5];
  const float* W2     = (const float*)d_in[6];
  const float* b2     = (const float*)d_in[7];

  int nin  = in_sizes[0] / 3;   // 60000
  int npts = nin * 8;           // 480000

  char* ws = (char*)d_ws;
  int*    map     = (int*)ws;                                    // 8 MiB (128^3)
  __bf16* featsbf = (__bf16*)(ws + ((size_t)8  << 20));          // 7.7 MB
  __bf16* x0      = (__bf16*)(ws + ((size_t)16 << 20));          // 61.4 MB
  __bf16* x1      = (__bf16*)(ws + ((size_t)80 << 20));          // 61.4 MB
  __bf16* Wupt    = (__bf16*)(ws + ((size_t)144 << 20));         // 64 KB
  __bf16* W1t     = Wupt + 8 * 4096;                             // 216 KB
  __bf16* W2t     = W1t + 27 * 4096;                             // 216 KB

  const int mapN = 128 * 128 * 128;
  k_init_map<<<(mapN + 255) / 256, 256, 0, stream>>>(map, mapN);
  k_build_map<<<(npts + 255) / 256, 256, 0, stream>>>(coords, map, npts);
  k_f32_to_bf16<<<(nin * 64 + 255) / 256, 256, 0, stream>>>(feats, featsbf, nin * 64);
  k_weight_tr<<<8, 256, 0, stream>>>(W_up, Wupt);
  k_weight_tr<<<27, 256, 0, stream>>>(W1, W1t);
  k_weight_tr<<<27, 256, 0, stream>>>(W2, W2t);

  int upwaves = ((nin + 15) / 16) * 8;
  k_upsample<<<(upwaves + 7) / 8, 256, 0, stream>>>(featsbf, Wupt, b_up, x0, nin);

  int ntile = (npts + 15) / 16;
  int cvblocks = (ntile + 31) / 32;
  k_conv<true,  false><<<cvblocks, 1024, CONV_LDS_BYTES, stream>>>(
      x0, W1t, b1, map, coords, x1, nullptr, npts);
  k_conv<false, true ><<<cvblocks, 1024, CONV_LDS_BYTES, stream>>>(
      x1, W2t, b2, map, coords, nullptr, (float*)d_out, npts);
}
